// MoE_44590350467674
// MI455X (gfx1250) — matrix operand order repaired and checked by reading, not compiled, not run on hardware
//
#include <hip/hip_runtime.h>
#include <hip/hip_bf16.h>

#define DI __device__ __forceinline__

typedef __bf16 bf16;
typedef __attribute__((ext_vector_type(16))) __bf16 v16bf;
typedef __attribute__((ext_vector_type(8)))  __bf16 v8bf;
typedef __attribute__((ext_vector_type(8)))  float  v8f;
typedef int v4i __attribute__((vector_size(16)));   // matches builtin param type

#define GLOBAL_AS __attribute__((address_space(1)))
#define LDS_AS    __attribute__((address_space(3)))

#if defined(__has_builtin)
#if __has_builtin(__builtin_amdgcn_global_load_async_to_lds_b128)
#define HAVE_ASYNC_LDS 1
#endif
#endif

namespace {
constexpr int kT   = 4096;   // B*S tokens
constexpr int kH   = 2048;   // hidden
constexpr int kD   = 1024;   // expert inner dim
constexpr int kERt = 16;     // routed experts
constexpr int kNE  = 18;     // 2 shared + 16 routed (unified expert list)
constexpr int BM = 128, BN = 128, BK = 32;
constexpr int THREADS = 256; // 8 waves (wave32), arranged 4x2, wave tile 32x64
}

// ---------------- WMMA helpers (CDNA5 bf16 16x16x32) ----------------

DI v8f wmma_bf16(v16bf a, v16bf b, v8f c) {
  return __builtin_amdgcn_wmma_f32_16x16x32_bf16(
      /*neg_a=*/false, a, /*neg_b=*/false, b,
      /*c_mod=*/(short)0, c, /*reuse_a=*/false, /*reuse_b=*/false);
}

// A fragment: 16x32 tile. lane (l&15) = row; (l>>4) selects K-halves {kh..kh+7, kh+16..kh+23}
DI v16bf load_a_frag(const bf16* lA, int mrow, int lane) {
  const int r  = mrow + (lane & 15);
  const int kh = (lane >> 4) * 8;
  const bf16* p = lA + r * BK;
  union { uint4 u; v8bf h; } lo, hi;
  lo.u = *(const uint4*)(p + kh);
  hi.u = *(const uint4*)(p + kh + 16);
  v16bf a;
#pragma unroll
  for (int i = 0; i < 8; ++i) { a[i] = lo.h[i]; a[i + 8] = hi.h[i]; }
  return a;
}

// B fragment: 32x16 tile, stored transposed in LDS as [n][k].
// lane (l&15) = column n; (l>>4) selects K-halves {kh..kh+7, kh+16..kh+23}, kh=(l>>4)*8.
DI v16bf load_b_frag(const bf16* lB, int ncol, int lane) {
  const int n = ncol + (lane & 15);
  const int g = (lane >> 4) * 8;
  const bf16* p = lB + n * BK;
  union { uint4 u; v8bf h; } lo, hi;
  lo.u = *(const uint4*)(p + g);
  hi.u = *(const uint4*)(p + g + 16);
  v16bf b;
#pragma unroll
  for (int i = 0; i < 8; ++i) { b[i] = lo.h[i]; b[i + 8] = hi.h[i]; }
  return b;
}

DI float gelu_exact(float x) {
  return 0.5f * x * (1.0f + erff(x * 0.70710678118654752f));
}

// 16B copy global -> LDS. Async direct-to-LDS on gfx1250 when available.
DI void cp16(bf16* ldst, const bf16* gsrc) {
#if HAVE_ASYNC_LDS
  __builtin_amdgcn_global_load_async_to_lds_b128(
      (GLOBAL_AS v4i*)(unsigned long long)(const void*)gsrc,
      (LDS_AS v4i*)(unsigned int)(unsigned long long)(void*)ldst,
      /*offset=*/0, /*cpol=*/0);
#else
  *(uint4*)ldst = *(const uint4*)gsrc;
#endif
}

DI void cp_commit() {
#if HAVE_ASYNC_LDS
#if __has_builtin(__builtin_amdgcn_s_wait_asynccnt)
  __builtin_amdgcn_s_wait_asynccnt(0);
#else
  asm volatile("s_wait_asynccnt 0x0" ::: "memory");
#endif
#endif
}

// ---------------- init kernels ----------------

__global__ void zero_out_kernel(float* __restrict__ out) {
  const int i = (blockIdx.x * blockDim.x + threadIdx.x) * 4;
  float4 z = {0.f, 0.f, 0.f, 0.f};
  *(float4*)(out + i) = z;
}

__global__ void route_init_kernel(int* __restrict__ cnt,
                                  int* __restrict__ idxl,
                                  float* __restrict__ gate) {
  const int t = blockIdx.x * blockDim.x + threadIdx.x;
  if (t < kNE) cnt[t] = (t < 2) ? kT : 0;   // shared experts see every token
  if (t < kT) {
    idxl[0 * kT + t] = t;
    idxl[1 * kT + t] = t;
    gate[0 * kT + t] = 1.0f;
    gate[1 * kT + t] = 1.0f;
  }
}

__global__ void convert_x_kernel(const float* __restrict__ x, bf16* __restrict__ xh) {
  const int i = (blockIdx.x * blockDim.x + threadIdx.x) * 4;
  float4 v = *(const float4*)(x + i);
  union { uint2 u; bf16 h[4]; } o;
  o.h[0] = (bf16)v.x; o.h[1] = (bf16)v.y; o.h[2] = (bf16)v.z; o.h[3] = (bf16)v.w;
  *(uint2*)(xh + i) = o.u;
}

// out[n][k] = (bf16) in[k][n], per expert (blockIdx.z). 32x32 LDS tiles.
__global__ void transpose_convert_kernel(const float* __restrict__ in,
                                         bf16* __restrict__ outT,
                                         int K, int N) {
  __shared__ float tile[32][33];
  const size_t base = (size_t)blockIdx.z * K * N;
  const int k0 = blockIdx.x * 32, n0 = blockIdx.y * 32;
  const int tx = threadIdx.x, ty = threadIdx.y;
#pragma unroll
  for (int j = 0; j < 32; j += 8)
    tile[ty + j][tx] = in[base + (size_t)(k0 + ty + j) * N + (n0 + tx)];
  __syncthreads();
  bf16* o = outT + base;
#pragma unroll
  for (int j = 0; j < 32; j += 8)
    o[(size_t)(n0 + ty + j) * K + (k0 + tx)] = (bf16)tile[tx][ty + j];
}

// ---------------- router: one wave32 per token ----------------

__global__ __launch_bounds__(32)
void router_kernel(const float* __restrict__ x,
                   const float* __restrict__ wr,   // [H][16]
                   int* __restrict__ cnt,
                   int* __restrict__ idxl,
                   float* __restrict__ gate) {
  const int t = blockIdx.x;
  const int lane = threadIdx.x;
  float acc[kERt];
#pragma unroll
  for (int e = 0; e < kERt; ++e) acc[e] = 0.0f;

  const float* xt = x + (size_t)t * kH;
  for (int h = lane; h < kH; h += 32) {
    const float xv = xt[h];
    const float4* w = (const float4*)(wr + (size_t)h * kERt);
    const float4 w0 = w[0], w1 = w[1], w2 = w[2], w3 = w[3];
    acc[0]  += xv * w0.x; acc[1]  += xv * w0.y; acc[2]  += xv * w0.z; acc[3]  += xv * w0.w;
    acc[4]  += xv * w1.x; acc[5]  += xv * w1.y; acc[6]  += xv * w1.z; acc[7]  += xv * w1.w;
    acc[8]  += xv * w2.x; acc[9]  += xv * w2.y; acc[10] += xv * w2.z; acc[11] += xv * w2.w;
    acc[12] += xv * w3.x; acc[13] += xv * w3.y; acc[14] += xv * w3.z; acc[15] += xv * w3.w;
  }
#pragma unroll
  for (int off = 16; off > 0; off >>= 1)
#pragma unroll
    for (int e = 0; e < kERt; ++e) acc[e] += __shfl_xor(acc[e], off, 32);

  // softmax (all lanes hold identical values)
  float mx = acc[0];
#pragma unroll
  for (int e = 1; e < kERt; ++e) mx = fmaxf(mx, acc[e]);
  float p[kERt]; float s = 0.0f;
#pragma unroll
  for (int e = 0; e < kERt; ++e) { p[e] = __expf(acc[e] - mx); s += p[e]; }
  const float inv = 1.0f / s;
#pragma unroll
  for (int e = 0; e < kERt; ++e) p[e] *= inv;

  // top-2 (first max wins ties, matching lax.top_k)
  int i1 = 0; float v1 = p[0];
#pragma unroll
  for (int e = 1; e < kERt; ++e) if (p[e] > v1) { v1 = p[e]; i1 = e; }
  int i2 = (i1 == 0) ? 1 : 0; float v2 = p[i2];
#pragma unroll
  for (int e = 0; e < kERt; ++e) if (e != i1 && p[e] > v2) { v2 = p[e]; i2 = e; }

  if (lane == 0) {
    int pos = atomicAdd(&cnt[2 + i1], 1);
    idxl[(2 + i1) * kT + pos] = t;
    gate[(2 + i1) * kT + pos] = v1;
    pos = atomicAdd(&cnt[2 + i2], 1);
    idxl[(2 + i2) * kT + pos] = t;
    gate[(2 + i2) * kT + pos] = v2;
  }
}

// ---------------- grouped up-GEMM: mid = gelu(gather(x) @ Wup) * gate ----------------

__global__ __launch_bounds__(THREADS)
void moe_up_kernel(const bf16* __restrict__ xh,
                   const bf16* __restrict__ wshUpT,   // [2][D][H]
                   const bf16* __restrict__ wrtUpT,   // [16][D][H]
                   const int*  __restrict__ idxl,
                   const float* __restrict__ gate,
                   const int*  __restrict__ cnt,
                   bf16* __restrict__ mid) {          // [18][T][D]
  const int e = blockIdx.z;
  const int count = cnt[e];
  const int m0 = blockIdx.y * BM;
  if (m0 >= count) return;
  const int n0 = blockIdx.x * BN;
  const bf16* WT = (e < 2) ? (wshUpT + (size_t)e * kD * kH)
                           : (wrtUpT + (size_t)(e - 2) * kD * kH);

  __shared__ __align__(32) bf16 lA[2][BM * BK];
  __shared__ __align__(32) bf16 lB[2][BN * BK];

  const int tid  = threadIdx.x;
  const int lane = tid & 31;
  const int wave = tid >> 5;
  const int wm = (wave >> 1) * 32;   // 4 wave-rows
  const int wn = (wave & 1) * 64;    // 2 wave-cols

  // staging: 512 16B chunks per tile, 2 per thread; gather rows fixed across K
  const int ac  = (tid & 3) * 8;
  const int ar0 = tid >> 2;
  const int ar1 = (tid + THREADS) >> 2;
  int s0 = m0 + ar0; if (s0 >= count) s0 = count - 1;
  int s1 = m0 + ar1; if (s1 >= count) s1 = count - 1;
  const bf16* aSrc0 = xh + (size_t)idxl[e * kT + s0] * kH + ac;
  const bf16* aSrc1 = xh + (size_t)idxl[e * kT + s1] * kH + ac;
  const int br0 = tid >> 2;
  const int br1 = (tid + THREADS) >> 2;
  const bf16* bSrc0 = WT + (size_t)(n0 + br0) * kH + ac;
  const bf16* bSrc1 = WT + (size_t)(n0 + br1) * kH + ac;

  auto stage = [&](int kb, int buf) {
    cp16(lA[buf] + ar0 * BK + ac, aSrc0 + kb * BK);
    cp16(lA[buf] + ar1 * BK + ac, aSrc1 + kb * BK);
    cp16(lB[buf] + br0 * BK + ac, bSrc0 + kb * BK);
    cp16(lB[buf] + br1 * BK + ac, bSrc1 + kb * BK);
  };

  v8f c0[4] = {}, c1[4] = {};

  const int ksteps = kH / BK;
  stage(0, 0);
  for (int kb = 0; kb < ksteps; ++kb) {
    const int buf = kb & 1;
    cp_commit();        // my async writes for stage kb are done
    __syncthreads();    // everyone's stage kb visible; prior reads of buf^1 done
    if (kb + 1 < ksteps) {
      stage(kb + 1, buf ^ 1);
      __builtin_prefetch(aSrc0 + (kb + 2) * BK, 0, 1);
      __builtin_prefetch(bSrc0 + (kb + 2) * BK, 0, 1);
    }
    const v16bf a0 = load_a_frag(lA[buf], wm,      lane);
    const v16bf a1 = load_a_frag(lA[buf], wm + 16, lane);
#pragma unroll
    for (int nn = 0; nn < 4; ++nn) {
      const v16bf b = load_b_frag(lB[buf], wn + nn * 16, lane);
      c0[nn] = wmma_bf16(a0, b, c0[nn]);
      c1[nn] = wmma_bf16(a1, b, c1[nn]);
    }
  }

  // epilogue: gelu * gate -> bf16 mid  (C layout: row=(l>>4)*8+j, col=l&15)
  const int cr = (lane >> 4) * 8;
  const int cc = lane & 15;
#pragma unroll
  for (int half = 0; half < 2; ++half) {
    const int mb = m0 + wm + half * 16 + cr;
#pragma unroll
    for (int j = 0; j < 8; ++j) {
      const int m = mb + j;
      if (m < count) {
        const float g = gate[e * kT + m];
        bf16* row = mid + ((size_t)e * kT + m) * kD + n0 + wn;
#pragma unroll
        for (int nn = 0; nn < 4; ++nn) {
          const float v = half ? c1[nn][j] : c0[nn][j];
          row[nn * 16 + cc] = (bf16)(gelu_exact(v) * g);
        }
      }
    }
  }
}

// ---------------- grouped down-GEMM: out[tok] += mid @ Wdown (scatter) ----------------

__global__ __launch_bounds__(THREADS)
void moe_down_kernel(const bf16* __restrict__ mid,      // [18][T][D]
                     const bf16* __restrict__ wshDownT, // [2][H][D]
                     const bf16* __restrict__ wrtDownT, // [16][H][D]
                     const int*  __restrict__ idxl,
                     const int*  __restrict__ cnt,
                     float* __restrict__ out) {         // [T][H]
  const int e = blockIdx.z;
  const int count = cnt[e];
  const int m0 = blockIdx.y * BM;
  if (m0 >= count) return;
  const int n0 = blockIdx.x * BN;
  const bf16* WT = (e < 2) ? (wshDownT + (size_t)e * kH * kD)
                           : (wrtDownT + (size_t)(e - 2) * kH * kD);

  __shared__ __align__(32) bf16 lA[2][BM * BK];
  __shared__ __align__(32) bf16 lB[2][BN * BK];

  const int tid  = threadIdx.x;
  const int lane = tid & 31;
  const int wave = tid >> 5;
  const int wm = (wave >> 1) * 32;
  const int wn = (wave & 1) * 64;

  const int ac  = (tid & 3) * 8;
  const int ar0 = tid >> 2;
  const int ar1 = (tid + THREADS) >> 2;
  const bf16* aSrc0 = mid + ((size_t)e * kT + m0 + ar0) * kD + ac;
  const bf16* aSrc1 = mid + ((size_t)e * kT + m0 + ar1) * kD + ac;
  const int br0 = tid >> 2;
  const int br1 = (tid + THREADS) >> 2;
  const bf16* bSrc0 = WT + (size_t)(n0 + br0) * kD + ac;
  const bf16* bSrc1 = WT + (size_t)(n0 + br1) * kD + ac;

  auto stage = [&](int kb, int buf) {
    cp16(lA[buf] + ar0 * BK + ac, aSrc0 + kb * BK);
    cp16(lA[buf] + ar1 * BK + ac, aSrc1 + kb * BK);
    cp16(lB[buf] + br0 * BK + ac, bSrc0 + kb * BK);
    cp16(lB[buf] + br1 * BK + ac, bSrc1 + kb * BK);
  };

  v8f c0[4] = {}, c1[4] = {};

  const int ksteps = kD / BK;
  stage(0, 0);
  for (int kb = 0; kb < ksteps; ++kb) {
    const int buf = kb & 1;
    cp_commit();
    __syncthreads();
    if (kb + 1 < ksteps) {
      stage(kb + 1, buf ^ 1);
      __builtin_prefetch(aSrc0 + (kb + 2) * BK, 0, 1);
      __builtin_prefetch(bSrc0 + (kb + 2) * BK, 0, 1);
    }
    const v16bf a0 = load_a_frag(lA[buf], wm,      lane);
    const v16bf a1 = load_a_frag(lA[buf], wm + 16, lane);
#pragma unroll
    for (int nn = 0; nn < 4; ++nn) {
      const v16bf b = load_b_frag(lB[buf], wn + nn * 16, lane);
      c0[nn] = wmma_bf16(a0, b, c0[nn]);
      c1[nn] = wmma_bf16(a1, b, c1[nn]);
    }
  }

  const int cr = (lane >> 4) * 8;
  const int cc = lane & 15;
#pragma unroll
  for (int half = 0; half < 2; ++half) {
    const int mb = m0 + wm + half * 16 + cr;
#pragma unroll
    for (int j = 0; j < 8; ++j) {
      const int m = mb + j;
      if (m < count) {
        const int tok = idxl[e * kT + m];
        float* row = out + (size_t)tok * kH + n0 + wn;
#pragma unroll
        for (int nn = 0; nn < 4; ++nn)
          atomicAdd(row + nn * 16 + cc, half ? c1[nn][j] : c0[nn][j]);
      }
    }
  }
}

// ---------------- launch ----------------

extern "C" void kernel_launch(void* const* d_in, const int* in_sizes, int n_in,
                              void* d_out, int out_size, void* d_ws, size_t ws_size,
                              hipStream_t stream) {
  const float* x        = (const float*)d_in[0];
  const float* Wsh_up   = (const float*)d_in[1];
  const float* Wsh_down = (const float*)d_in[2];
  const float* Wrt_up   = (const float*)d_in[3];
  const float* Wrt_down = (const float*)d_in[4];
  const float* Wrouter  = (const float*)d_in[5];
  float* out = (float*)d_out;

  char* p = (char*)d_ws;
  auto alloc = [&](size_t bytes) -> void* {
    void* r = (void*)p;
    p += (bytes + 255) & ~(size_t)255;
    return r;
  };
  bf16* xh       = (bf16*)alloc((size_t)kT * kH * sizeof(bf16));
  bf16* wshUpT   = (bf16*)alloc((size_t)2  * kD * kH * sizeof(bf16));
  bf16* wrtUpT   = (bf16*)alloc((size_t)16 * kD * kH * sizeof(bf16));
  bf16* wshDownT = (bf16*)alloc((size_t)2  * kH * kD * sizeof(bf16));
  bf16* wrtDownT = (bf16*)alloc((size_t)16 * kH * kD * sizeof(bf16));
  float* gate    = (float*)alloc((size_t)kNE * kT * sizeof(float));
  int*   idxl    = (int*)alloc((size_t)kNE * kT * sizeof(int));
  int*   cnt     = (int*)alloc(256);
  bf16*  mid     = (bf16*)alloc((size_t)kNE * kT * kD * sizeof(bf16));

  zero_out_kernel<<<(kT * kH / 4) / 256, 256, 0, stream>>>(out);
  route_init_kernel<<<kT / 256, 256, 0, stream>>>(cnt, idxl, gate);
  convert_x_kernel<<<(kT * kH / 4) / 256, 256, 0, stream>>>(x, xh);
  // up weights: in [H][D] -> out [D][H]
  transpose_convert_kernel<<<dim3(kH / 32, kD / 32, 2),  dim3(32, 8), 0, stream>>>(Wsh_up,   wshUpT,   kH, kD);
  transpose_convert_kernel<<<dim3(kH / 32, kD / 32, 16), dim3(32, 8), 0, stream>>>(Wrt_up,   wrtUpT,   kH, kD);
  // down weights: in [D][H] -> out [H][D]
  transpose_convert_kernel<<<dim3(kD / 32, kH / 32, 2),  dim3(32, 8), 0, stream>>>(Wsh_down, wshDownT, kD, kH);
  transpose_convert_kernel<<<dim3(kD / 32, kH / 32, 16), dim3(32, 8), 0, stream>>>(Wrt_down, wrtDownT, kD, kH);

  router_kernel<<<kT, 32, 0, stream>>>(x, Wrouter, cnt, idxl, gate);

  moe_up_kernel<<<dim3(kD / BN, kT / BM, kNE), THREADS, 0, stream>>>(
      xh, wshUpT, wrtUpT, idxl, gate, cnt, mid);
  moe_down_kernel<<<dim3(kH / BN, kT / BM, kNE), THREADS, 0, stream>>>(
      mid, wshDownT, wrtDownT, idxl, cnt, out);
}
